// SE3TransformerWadjacency_31129922961612
// MI455X (gfx1250) — matrix operand order repaired and checked by reading, not compiled, not run on hardware
//
#include <hip/hip_runtime.h>
#include <math.h>

#define DEV __device__ __forceinline__

constexpr int B_  = 8;
constexpr int N_  = 8192;
constexpr int K_  = 4;
constexpr int H_  = 4;
constexpr int D_  = 16;
constexpr int L_  = 4;
constexpr int E_  = 128;
constexpr int TC_ = 61;
constexpr float EPS_ = 1e-6f;

typedef __attribute__((ext_vector_type(16))) __bf16          v16bf;
typedef __attribute__((ext_vector_type(8)))  float           v8f;
typedef __attribute__((ext_vector_type(8)))  unsigned short  us8;
typedef __attribute__((ext_vector_type(16))) unsigned short  us16;
typedef __attribute__((ext_vector_type(4)))  unsigned int    u32x4;
typedef __attribute__((ext_vector_type(8)))  int             i32x8;
typedef __attribute__((ext_vector_type(4)))  int             i32x4;

DEV unsigned short f2bf(float f){
  unsigned u = __builtin_bit_cast(unsigned, f);
  u += 0x7fffu + ((u >> 16) & 1u);          // round-to-nearest-even
  return (unsigned short)(u >> 16);
}

DEV v16bf load_frag16(const unsigned short* p0, const unsigned short* p1){
  union { us16 w; us8 h[2]; } u;
  u.h[0] = *(const us8*)p0;                  // b128 load (global or ds)
  u.h[1] = *(const us8*)p1;
  return __builtin_bit_cast(v16bf, u.w);
}

DEV float fast_silu(float v){
  return v * __builtin_amdgcn_rcpf(1.f + __expf(-v));
}

// ---------------------------------------------------------------------------
// Generic bf16 WMMA GEMM: C[M x NOUT] (+)= A[M x KDIM] * Bt[NOUT x KDIM]^T
// Bt is pre-transposed [col][k] and staged into LDS via the Tensor Data Mover.
// FLAGS: 1=residual add into f32 C, 2=SiLU, 4=store f32 C, 8=store bf16 Ob
// ---------------------------------------------------------------------------
template<int KDIM, int NOUT, int FLAGS>
__global__ __launch_bounds__(256) void gemm_bf16(
    const unsigned short* __restrict__ A,
    const unsigned short* __restrict__ Bt,
    float* __restrict__ C, int ldc,
    unsigned short* __restrict__ Ob, int ldob)
{
  constexpr int KC = KDIM / 32;
  constexpr int NT = NOUT / 16;
  __shared__ unsigned short sB[NOUT * KDIM];

#if __has_builtin(__builtin_amdgcn_tensor_load_to_lds) && \
    __has_builtin(__builtin_amdgcn_s_wait_tensorcnt)
  // --- TDM: one wave DMAs the whole weight tile (NOUT x KDIM bf16) to LDS ---
  if (threadIdx.x < 32){
    const unsigned long long ga = (unsigned long long)(const void*)Bt;
    const unsigned ldsOff = (unsigned)(unsigned long long)(void*)&sB[0]; // LDS byte address, low 32 bits
    u32x4 g0;
    g0[0] = 1u;                                            // count=1 (valid user D#)
    g0[1] = ldsOff;                                        // lds_addr
    g0[2] = (unsigned)(ga & 0xffffffffull);                // global_addr[31:0]
    g0[3] = (unsigned)((ga >> 32) & 0x01ffffffull)         // global_addr[56:32]
          | (2u << 30);                                    // type=2 ("image")
    i32x8 g1;
    g1[0] = (int)(1u << 16);                               // data_size=1 -> 2 bytes
    g1[1] = (int)((KDIM & 0xffff) << 16);                  // tensor_dim0[15:0]
    g1[2] = (int)(((unsigned)KDIM >> 16) | ((NOUT & 0xffff) << 16)); // dim0 hi | dim1 lo
    g1[3] = (int)(((unsigned)NOUT >> 16) | ((KDIM & 0xffff) << 16)); // dim1 hi | tile_dim0
    g1[4] = (int)(NOUT & 0xffff);                          // tile_dim1 (tile_dim2=0)
    g1[5] = (int)KDIM;                                     // tensor_dim0_stride[31:0]
    g1[6] = 0;
    g1[7] = 0;
    i32x4 gz4 = {0,0,0,0};                                 // 2-D tensor: groups 2/3 unused
    i32x8 gz8 = {0,0,0,0,0,0,0,0};                         // extra group (clang-23 6-arg form)
    __builtin_amdgcn_tensor_load_to_lds(g0, g1, gz4, gz4, gz8, 0);
    __builtin_amdgcn_s_wait_tensorcnt(0);
  }
#else
  for (int i = threadIdx.x; i < NOUT*KDIM/8; i += 256)
    ((us8*)sB)[i] = ((const us8*)Bt)[i];
#endif
  __syncthreads();

  const int lane = threadIdx.x & 31;
  const int wave = threadIdx.x >> 5;
  const int li = lane & 15;
  const int hi = lane >> 4;
  const long rowBase = (long)blockIdx.x * 128 + wave * 16;

  // A fragment per ISA 16-bit A 16x32 layout: element j ->
  // K = kc*32 + (j/8)*16 + hi*8 + (j%8)  => two contiguous 16B global loads
  const unsigned short* ap = A + (rowBase + li) * KDIM;
  v16bf afr[KC];
#pragma unroll
  for (int kc = 0; kc < KC; ++kc){
    const unsigned short* p = ap + kc*32 + hi*8;
    afr[kc] = load_frag16(p, p + 16);
  }

#pragma unroll
  for (int nt = 0; nt < NT; ++nt){
    v8f acc = {0.f,0.f,0.f,0.f,0.f,0.f,0.f,0.f};
    const int col = nt*16 + li;
    // B fragment from LDS: lane = column, element j -> K = kc*32 + (j/8)*16 + hi*8 + (j%8)
    const unsigned short* bp = sB + col*KDIM + hi*8;
#pragma unroll
    for (int kc = 0; kc < KC; ++kc){
      v16bf bfr = load_frag16(bp + kc*32, bp + kc*32 + 16);
      acc = __builtin_amdgcn_wmma_f32_16x16x32_bf16(
                false, afr[kc], false, bfr, (short)0, acc, false, false);
    }
    // D layout: VGPR r -> row hi*8+r, lane li -> col
    const long r0 = rowBase + hi*8;
#pragma unroll
    for (int r = 0; r < 8; ++r){
      float v = acc[r];
      const long row = r0 + r;
      if constexpr (FLAGS & 1) v += C[row*ldc + col];
      if constexpr (FLAGS & 2) v = fast_silu(v);
      if constexpr (FLAGS & 4) C[row*ldc + col] = v;
      if constexpr (FLAGS & 8) Ob[row*ldob + col] = f2bf(v);
    }
  }
}

// ---------------------------------------------------------------------------
// Fused LayerNorm(f0)->g0(bf16) and vnorm(f1)->g1(bf16).  f1 layout [b][n][v][c]
// ---------------------------------------------------------------------------
__global__ __launch_bounds__(256) void norm_kernel(
    const float* __restrict__ f0, const float* __restrict__ f1,
    unsigned short* __restrict__ g0, unsigned short* __restrict__ g1)
{
  const long i = (long)blockIdx.x*256 + threadIdx.x;   // b*N + n
  const float* p0 = f0 + i*64;
  float s = 0.f, ss = 0.f;
  for (int c = 0; c < 64; ++c){ float v = p0[c]; s += v; ss += v*v; }
  const float mu  = s * (1.f/64.f);
  const float var = ss * (1.f/64.f) - mu*mu;
  const float rs  = rsqrtf(var + EPS_);
  unsigned short* q0 = g0 + i*64;
  for (int c = 0; c < 64; ++c) q0[c] = f2bf((p0[c]-mu)*rs);

  const float* p1 = f1 + i*192;
  float s1 = 0.f;
  for (int c = 0; c < 192; ++c){ float v = p1[c]; s1 += v*v; }
  const float sv = rsqrtf(s1 * (1.f/64.f) + EPS_);    // mean over C of sum_v sq
  unsigned short* q1 = g1 + i*192;
  for (int c = 0; c < 192; ++c) q1[c] = f2bf(p1[c]*sv);
}

// ---------------------------------------------------------------------------
// Attention: K=4 neighbors, per-thread = (b,n,h).  qkv layout per row:
// [q(64) | k(64) | v(64)].  Neighbor k/v gathered from the per-node arrays.
// ---------------------------------------------------------------------------
__global__ __launch_bounds__(256) void attn_kernel(
    const float* __restrict__ qkv0, const float* __restrict__ qkv1,
    const float* __restrict__ rhat, const float* __restrict__ mbias,
    const int*  __restrict__ nbr,
    unsigned short* __restrict__ o0, unsigned short* __restrict__ o1)
{
  const int tid = blockIdx.x*256 + threadIdx.x;        // bn*H + h
  const int h   = tid & (H_-1);
  const long bn = tid >> 2;
  const int n   = (int)(bn & (N_-1));
  const int b   = (int)(bn >> 13);

  float q0r[D_], q1r[3][D_];
  {
    const float* q0p = qkv0 + bn*192 + h*D_;
#pragma unroll
    for (int d = 0; d < D_; ++d) q0r[d] = q0p[d];
#pragma unroll
    for (int v = 0; v < 3; ++v){
      const float* q1p = qkv1 + (bn*3 + v)*192 + h*D_;
      for (int d = 0; d < D_; ++d) q1r[v][d] = q1p[d];
    }
  }

  int jj[K_]; float lg[K_];
#pragma unroll
  for (int k = 0; k < K_; ++k){
    const int j = nbr[n*K_ + k]; jj[k] = j;
    const long jb = (long)b*N_ + j;
    const float* k0p = qkv0 + jb*192 + 64 + h*D_;
    float acc = 0.f;
    for (int d = 0; d < D_; ++d) acc += q0r[d]*k0p[d];
#pragma unroll
    for (int v = 0; v < 3; ++v){
      const float* k1p = qkv1 + (jb*3 + v)*192 + 64 + h*D_;
      for (int d = 0; d < D_; ++d) acc += q1r[v][d]*k1p[d];
    }
    lg[k] = acc * 0.25f + mbias[n*K_ + k];             // scale = D^-0.5
  }
  float mx = lg[0];
  for (int k = 1; k < K_; ++k) mx = fmaxf(mx, lg[k]);
  float aw[K_], den = 0.f;
  for (int k = 0; k < K_; ++k){ aw[k] = __expf(lg[k]-mx); den += aw[k]; }
  const float inv = __builtin_amdgcn_rcpf(den);
  for (int k = 0; k < K_; ++k) aw[k] *= inv;

  float o0r[D_] = {};
  float o1r[3][D_] = {};
#pragma unroll
  for (int k = 0; k < K_; ++k){
    const long jb = (long)b*N_ + jj[k];
    const float* rh = rhat + (bn*K_ + k)*3;
    const float r0 = rh[0], r1 = rh[1], r2 = rh[2];
    const float a  = aw[k];
    const float* v0p = qkv0 + jb*192 + 128 + h*D_;
    const float* v1x = qkv1 + (jb*3 + 0)*192 + 128 + h*D_;
    const float* v1y = qkv1 + (jb*3 + 1)*192 + 128 + h*D_;
    const float* v1z = qkv1 + (jb*3 + 2)*192 + 128 + h*D_;
    for (int d = 0; d < D_; ++d){
      const float v0 = v0p[d];
      const float x1 = v1x[d], y1 = v1y[d], z1 = v1z[d];
      o0r[d]    += a * (v0 + x1*r0 + y1*r1 + z1*r2);   // v0 + v1.rhat
      o1r[0][d] += a * (x1 + v0*r0);                   // v1 + v0*rhat
      o1r[1][d] += a * (y1 + v0*r1);
      o1r[2][d] += a * (z1 + v0*r2);
    }
  }
  unsigned short* po0 = o0 + bn*64 + h*D_;
  for (int d = 0; d < D_; ++d) po0[d] = f2bf(o0r[d]);
#pragma unroll
  for (int v = 0; v < 3; ++v){
    unsigned short* po1 = o1 + (bn*3 + v)*64 + h*D_;
    for (int d = 0; d < D_; ++d) po1[d] = f2bf(o1r[v][d]);
  }
}

// ---------------------------------------------------------------------------
// Time-embedding MLP: te(B,128)->silu(@t_w1+b1)->@t_w2+b2 -> (B,61)
// ---------------------------------------------------------------------------
__global__ void time_mlp_kernel(const float* __restrict__ t,
    const float* __restrict__ w1, const float* __restrict__ b1,
    const float* __restrict__ w2, const float* __restrict__ b2,
    float* __restrict__ te)
{
  __shared__ float sh[E_];
  const int b = blockIdx.x;
  const int e = threadIdx.x;
  const float tv = t[b];
  const int half = E_/2;
  const int idx = (e < half) ? e : (e - half);
  const float fr = __expf(-logf(10000.f) * (float)idx / (float)half);
  const float ph = tv * fr;
  sh[e] = (e < half) ? __sinf(ph) : __cosf(ph);
  __syncthreads();
  float acc = b1[e];
  for (int i = 0; i < E_; ++i) acc += sh[i] * w1[i*E_ + e];
  const float hid = fast_silu(acc);
  __syncthreads();
  sh[e] = hid;
  __syncthreads();
  if (e < TC_){
    float a2 = b2[e];
    for (int i = 0; i < E_; ++i) a2 += sh[i] * w2[i*TC_ + e];
    te[b*TC_ + e] = a2;
  }
}

__global__ __launch_bounds__(256) void init_f_kernel(
    const float* __restrict__ x, const float* __restrict__ win1,
    const float* __restrict__ te, float* __restrict__ f0, float* __restrict__ f1)
{
  const long i = (long)blockIdx.x*256 + threadIdx.x;   // b*N+n
  const int b = (int)(i >> 13);
  const float* tp = te + b*TC_;
  float* F0 = f0 + i*64;
  for (int c = 0; c < TC_; ++c) F0[c] = tp[c];
  float xv[3];
#pragma unroll
  for (int v = 0; v < 3; ++v){ xv[v] = x[i*3 + v]; F0[TC_ + v] = xv[v]; }
  float* F1 = f1 + i*192;
#pragma unroll
  for (int v = 0; v < 3; ++v)
    for (int c = 0; c < 64; ++c) F1[v*64 + c] = xv[v] * win1[c];
}

__global__ __launch_bounds__(256) void rhat_kernel(
    const float* __restrict__ x, const int* __restrict__ nbr,
    const int* __restrict__ msk, float* __restrict__ rhat,
    float* __restrict__ mbias)
{
  const int i = blockIdx.x*256 + threadIdx.x;          // bn*K + k
  const int k  = i & (K_-1);
  const long bn = i >> 2;
  const int n  = (int)(bn & (N_-1));
  const int b  = (int)(bn >> 13);
  const int j  = nbr[n*K_ + k];
  const long pj = ((long)b*N_ + j)*3, pn = bn*3;
  const float dx = x[pj+0]-x[pn+0], dy = x[pj+1]-x[pn+1], dz = x[pj+2]-x[pn+2];
  const float rr = rsqrtf(dx*dx + dy*dy + dz*dz + EPS_);
  float* rh = rhat + (bn*K_ + k)*3;
  rh[0] = dx*rr; rh[1] = dy*rr; rh[2] = dz*rr;
  if (b == 0) mbias[n*K_ + k] = msk[n*K_ + k] ? 0.f : -1e9f;
}

__global__ __launch_bounds__(256) void score_kernel(
    const float* __restrict__ f1, const float* __restrict__ Wout,
    const float* __restrict__ bout, float* __restrict__ out)
{
  const int tid = blockIdx.x*256 + threadIdx.x;        // bn*3 + v
  const int v = tid % 3;
  const long bn = tid / 3;
  const float* p = f1 + (bn*3 + v)*64;
  float acc = bout[v];
  for (int c = 0; c < 64; ++c) acc += p[c] * Wout[v*64 + c];
  out[tid] = acc;
}

// ---------------------------------------------------------------------------
// Weight prep: convert f32 -> bf16, transpose to [out][k].
// Per-layer block (69632 bf16): qkv0@0, qkv1@12288, o0@24576, o1@28672,
// ff0a@32768, ff0b@49152, ff1@65536.
// ---------------------------------------------------------------------------
__global__ __launch_bounds__(256) void prep_weights_kernel(
    const float* Wq0, const float* Wk0, const float* Wv0,
    const float* Wq1, const float* Wk1, const float* Wv1,
    const float* Wo0, const float* Wo1,
    const float* Wff0a, const float* Wff0b, const float* Wff1,
    unsigned short* wb)
{
  const int tid = blockIdx.x*256 + threadIdx.x;
  if (tid >= L_*69632) return;
  const int l = tid / 69632;
  const int r = tid - l*69632;
  float v;
  if (r < 12288){                                        // qkv0 [192][64]
    const int e = r/64, c = r%64, sel = e/64, ee = e%64;
    const float* W = (sel==0)?Wq0:(sel==1)?Wk0:Wv0;
    v = W[l*4096 + c*64 + ee];
  } else if (r < 24576){                                 // qkv1 [192][64]
    const int rr = r-12288, e = rr/64, c = rr%64, sel = e/64, ee = e%64;
    const float* W = (sel==0)?Wq1:(sel==1)?Wk1:Wv1;
    v = W[l*4096 + c*64 + ee];
  } else if (r < 28672){                                 // o0 [64 c][64 e]
    const int rr = r-24576, c = rr/64, e = rr%64;
    v = Wo0[l*4096 + e*64 + c];
  } else if (r < 32768){                                 // o1 [64 c][64 e]
    const int rr = r-28672, c = rr/64, e = rr%64;
    v = Wo1[l*4096 + e*64 + c];
  } else if (r < 49152){                                 // ff0a [256 o][64 c]
    const int rr = r-32768, o = rr/64, c = rr%64;
    v = Wff0a[l*16384 + c*256 + o];
  } else if (r < 65536){                                 // ff0b [64 o][256 k]
    const int rr = r-49152, o = rr/256, kk = rr%256;
    v = Wff0b[l*16384 + kk*64 + o];
  } else {                                               // ff1 [64][64]
    const int rr = r-65536, e = rr/64, c = rr%64;
    v = Wff1[l*4096 + c*64 + e];
  }
  wb[tid] = f2bf(v);
}

// ---------------------------------------------------------------------------
extern "C" void kernel_launch(void* const* d_in, const int* in_sizes, int n_in,
                              void* d_out, int out_size, void* d_ws, size_t ws_size,
                              hipStream_t stream)
{
  (void)in_sizes; (void)n_in; (void)out_size; (void)ws_size;
  const float* x     = (const float*)d_in[0];
  /* d_in[1] (y) unused by the reference */
  const float* t     = (const float*)d_in[2];
  const int*   nbr   = (const int*)  d_in[3];
  const int*   msk   = (const int*)  d_in[4];
  const float* t_w1  = (const float*)d_in[5];
  const float* t_b1  = (const float*)d_in[6];
  const float* t_w2  = (const float*)d_in[7];
  const float* t_b2  = (const float*)d_in[8];
  const float* win1  = (const float*)d_in[9];
  const float* Wq0   = (const float*)d_in[10];
  const float* Wk0   = (const float*)d_in[11];
  const float* Wv0   = (const float*)d_in[12];
  const float* Wq1   = (const float*)d_in[13];
  const float* Wk1   = (const float*)d_in[14];
  const float* Wv1   = (const float*)d_in[15];
  const float* Wo0   = (const float*)d_in[16];
  const float* Wo1   = (const float*)d_in[17];
  const float* Wff0a = (const float*)d_in[18];
  const float* Wff0b = (const float*)d_in[19];
  const float* Wff1  = (const float*)d_in[20];
  const float* Wout  = (const float*)d_in[21];
  const float* bout  = (const float*)d_in[22];

  char* ws8 = (char*)d_ws;
  size_t off = 0;
  auto take = [&](size_t bytes)->void*{
    off = (off + 255) & ~(size_t)255;
    void* p = ws8 + off;
    off += bytes;
    return p;
  };
  const size_t BN  = (size_t)B_*N_;
  float* f0    = (float*) take(BN*64*4);
  float* f1    = (float*) take(BN*192*4);
  unsigned short* g0 = (unsigned short*) take(BN*64*2);    // aliased: o0
  unsigned short* g1 = (unsigned short*) take(BN*192*2);   // aliased: o1
  float* qkv0  = (float*) take(BN*192*4);                  // aliased: ffn hidden u
  float* qkv1  = (float*) take(BN*3*192*4);
  float* rhat  = (float*) take(BN*K_*3*4);
  float* mbias = (float*) take((size_t)N_*K_*4);
  float* te    = (float*) take((size_t)B_*TC_*4);
  unsigned short* wb = (unsigned short*) take((size_t)L_*69632*2);
  unsigned short* o0 = g0;
  unsigned short* o1 = g1;
  unsigned short* ubuf = (unsigned short*)qkv0;

  prep_weights_kernel<<<(L_*69632 + 255)/256, 256, 0, stream>>>(
      Wq0, Wk0, Wv0, Wq1, Wk1, Wv1, Wo0, Wo1, Wff0a, Wff0b, Wff1, wb);
  time_mlp_kernel<<<B_, E_, 0, stream>>>(t, t_w1, t_b1, t_w2, t_b2, te);
  init_f_kernel<<<(int)(BN/256), 256, 0, stream>>>(x, win1, te, f0, f1);
  rhat_kernel<<<(int)(BN*K_/256), 256, 0, stream>>>(x, nbr, msk, rhat, mbias);

  const int gBN  = (int)(BN / 128);        // 512  (rows = B*N)
  const int gBN3 = (int)(BN*3 / 128);      // 1536 (rows = B*N*3)
  for (int l = 0; l < L_; ++l){
    unsigned short* wl = wb + (size_t)l*69632;
    norm_kernel<<<(int)(BN/256), 256, 0, stream>>>(f0, f1, g0, g1);
    // qkv projections (k/v computed per node, gathered later in attention)
    gemm_bf16< 64,192,4><<<gBN,  256, 0, stream>>>(g0, wl+0,     qkv0, 192, nullptr, 0);
    gemm_bf16< 64,192,4><<<gBN3, 256, 0, stream>>>(g1, wl+12288, qkv1, 192, nullptr, 0);
    attn_kernel<<<(int)(BN*H_/256), 256, 0, stream>>>(qkv0, qkv1, rhat, mbias, nbr, o0, o1);
    // output projections + residual
    gemm_bf16< 64, 64,5><<<gBN,  256, 0, stream>>>(o0, wl+24576, f0, 64, nullptr, 0);
    gemm_bf16< 64, 64,5><<<gBN3, 256, 0, stream>>>(o1, wl+28672, f1, 64, nullptr, 0);
    // FFN
    norm_kernel<<<(int)(BN/256), 256, 0, stream>>>(f0, f1, g0, g1);
    gemm_bf16< 64,256,10><<<gBN, 256, 0, stream>>>(g0, wl+32768, nullptr, 0, ubuf, 256);
    gemm_bf16<256, 64,5><<<gBN,  256, 0, stream>>>(ubuf, wl+49152, f0, 64, nullptr, 0);
    gemm_bf16< 64, 64,5><<<gBN3, 256, 0, stream>>>(g1, wl+65536, f1, 64, nullptr, 0);
  }
  score_kernel<<<(int)(BN*3/256), 256, 0, stream>>>(f1, Wout, bout, (float*)d_out);
}
